// GAT_Model_14594298872000
// MI455X (gfx1250) — compile-verified, matrix operand order checked by reading, not run on hardware
//
#include <hip/hip_runtime.h>

#define NN 1500
#define GAT_IN 32
#define GAT_OUT 8
#define MLP_HIDDEN 128
#define ALPHA_SLOPE 0.2f
#define NEG_INF_F -9000000000000000.0f

typedef __attribute__((ext_vector_type(16))) _Float16 v16h;
typedef __attribute__((ext_vector_type(8)))  float    v8f;

// ---------------------------------------------------------------------------
// GAT projection: h = enc @ W (K x 8), f_src = h @ a[:8], f_dst = h @ a[8:]
// One thread per node. K = 32 (layer 0) or 8 (layers 1,2). Tiny, VALU-bound.
// ---------------------------------------------------------------------------
__global__ __launch_bounds__(256) void gat_proj(
                         const float* __restrict__ enc, int K,
                         const float* __restrict__ W, const float* __restrict__ a,
                         float* __restrict__ h_out, float* __restrict__ fsrc,
                         float* __restrict__ fdst)
{
    int n = blockIdx.x * blockDim.x + threadIdx.x;
    if (n >= NN) return;
    float h[GAT_OUT];
#pragma unroll
    for (int o = 0; o < GAT_OUT; ++o) h[o] = 0.f;
    for (int k = 0; k < K; ++k) {
        float x = enc[n * K + k];
#pragma unroll
        for (int o = 0; o < GAT_OUT; ++o) h[o] += x * W[k * GAT_OUT + o];
    }
    float fs = 0.f, fd = 0.f;
#pragma unroll
    for (int o = 0; o < GAT_OUT; ++o) {
        h_out[n * GAT_OUT + o] = h[o];
        fs += h[o] * a[o];
        fd += h[o] * a[GAT_OUT + o];
    }
    fsrc[n] = fs;
    fdst[n] = fd;
}

// ---------------------------------------------------------------------------
// GAT attention row: one block (256 thr = 8 wave32) per row i.
//   e_ij = leaky_relu(fsrc_i + fdst_j) * (geo+sem)  where adj>0 else -inf
//   att  = softmax_j(e);  enc_out_i = elu(sum_j att_ij * h_j)
// Bandwidth-bound on the two 9MB adjacency reads. LDS row buffer (6KB) +
// wave32 shfl_xor reductions.
// ---------------------------------------------------------------------------
__global__ __launch_bounds__(256) void gat_attn(
                         const float* __restrict__ geo, const float* __restrict__ sem,
                         const float* __restrict__ h, const float* __restrict__ fsrc,
                         const float* __restrict__ fdst, float* __restrict__ enc_out)
{
    __shared__ float sE[NN];
    __shared__ float sWmax[8];
    __shared__ float sPart[8][GAT_OUT + 1];

    const int i    = blockIdx.x;
    const int tid  = threadIdx.x;
    const int wid  = tid >> 5;
    const int lane = tid & 31;
    const float fsi = fsrc[i];
    const long rowOff = (long)i * NN;

    // pass 1: scores + row max
    float lmax = NEG_INF_F;
    for (int j = tid; j < NN; j += 256) {
        float adj = geo[rowOff + j] + sem[rowOff + j];
        float ev = NEG_INF_F;
        if (adj > 0.f) {
            float s = fsi + fdst[j];
            s = (s > 0.f) ? s : ALPHA_SLOPE * s;
            ev = s * adj;
        }
        sE[j] = ev;
        lmax = fmaxf(lmax, ev);
    }
#pragma unroll
    for (int m = 16; m >= 1; m >>= 1) lmax = fmaxf(lmax, __shfl_xor(lmax, m, 32));
    if (lane == 0) sWmax[wid] = lmax;
    __syncthreads();
    float rowMax = sWmax[0];
#pragma unroll
    for (int w = 1; w < 8; ++w) rowMax = fmaxf(rowMax, sWmax[w]);

    // pass 2: exp-sum + weighted aggregation of h (2% density -> skip masked)
    float lsum = 0.f;
    float lacc[GAT_OUT];
#pragma unroll
    for (int o = 0; o < GAT_OUT; ++o) lacc[o] = 0.f;
    for (int j = tid; j < NN; j += 256) {
        float ev = sE[j];
        if (ev > -8.9e15f) {
            float p = __expf(ev - rowMax);
            lsum += p;
            const float* hj = h + j * GAT_OUT;
#pragma unroll
            for (int o = 0; o < GAT_OUT; ++o) lacc[o] += p * hj[o];
        }
    }
#pragma unroll
    for (int m = 16; m >= 1; m >>= 1) {
        lsum += __shfl_xor(lsum, m, 32);
#pragma unroll
        for (int o = 0; o < GAT_OUT; ++o) lacc[o] += __shfl_xor(lacc[o], m, 32);
    }
    if (lane == 0) {
        sPart[wid][0] = lsum;
#pragma unroll
        for (int o = 0; o < GAT_OUT; ++o) sPart[wid][1 + o] = lacc[o];
    }
    __syncthreads();
    if (tid == 0) {
        float tsum = 0.f;
        float tacc[GAT_OUT];
#pragma unroll
        for (int o = 0; o < GAT_OUT; ++o) tacc[o] = 0.f;
#pragma unroll
        for (int w = 0; w < 8; ++w) {
            tsum += sPart[w][0];
#pragma unroll
            for (int o = 0; o < GAT_OUT; ++o) tacc[o] += sPart[w][1 + o];
        }
        float inv = 1.f / tsum;
#pragma unroll
        for (int o = 0; o < GAT_OUT; ++o) {
            float v = tacc[o] * inv;
            enc_out[i * GAT_OUT + o] = (v > 0.f) ? v : (__expf(v) - 1.f);
        }
    }
}

// ---------------------------------------------------------------------------
// Fused edge-MLP with WMMA (v_wmma_f32_16x16x32_f16).
// Each wave owns 16-pair tiles: A = [16 x 32] (enc_i | enc_j | dist | pad),
// B chunks = fc1_w column tiles [32 x 16] built once per wave and kept
// RESIDENT in VGPRs -- __launch_bounds__(256, 1) raises the register budget
// so the 64 B-VGPRs do not spill to scratch (they did at default occupancy).
// 8 WMMAs -> full 128 hidden units; bias+relu+scale by fc2_w, then a 16-lane
// shfl reduction forms the final dot product. One f32 store per pair.
// ---------------------------------------------------------------------------
__global__ __launch_bounds__(256, 1) void edge_mlp_wmma(
                              const float* __restrict__ enc,
                              const float* __restrict__ dist,
                              const float* __restrict__ fc1_w,
                              const float* __restrict__ fc1_b,
                              const float* __restrict__ fc2_w,
                              const float* __restrict__ fc2_b,
                              float* __restrict__ out, int numTiles)
{
    const int tid  = threadIdx.x;
    const int lane = tid & 31;
    const int l    = lane & 15;   // row/col within 16
    const int hf   = lane >> 4;   // lane half selects K-group (ISA A/B layout)
    const int gwave  = (blockIdx.x * blockDim.x + tid) >> 5;
    const int nwaves = (gridDim.x * blockDim.x) >> 5;

    // Build B register tiles for all 8 hidden chunks (K = 0..16 valid, rest 0).
    // Lane layout (16-bit B, K=32): lanes 0-15 hold K{0-7,16-23}, lanes 16-31 K{8-15,24-31}.
    v16h Breg[8];
    float bias[8], w2[8];
#pragma unroll
    for (int c = 0; c < 8; ++c) {
        const int ncol = c * 16 + l;
        const int k0 = hf ? 8 : 0;
        const int k1 = hf ? 24 : 16;
        v16h b;
#pragma unroll
        for (int t = 0; t < 8; ++t) {
            int k = k0 + t;
            b[t] = (k < 17) ? (_Float16)fc1_w[k * MLP_HIDDEN + ncol] : (_Float16)0.f;
        }
#pragma unroll
        for (int t = 0; t < 8; ++t) {
            int k = k1 + t;
            b[8 + t] = (k < 17) ? (_Float16)fc1_w[k * MLP_HIDDEN + ncol] : (_Float16)0.f;
        }
        Breg[c] = b;
        bias[c] = fc1_b[ncol];
        w2[c]   = fc2_w[ncol];
    }
    const float b2 = fc2_b[0];

    for (int tile = gwave; tile < numTiles; tile += nwaves) {
        const int p = tile * 16 + l;          // pair handled by this lane group
        const int i = p / NN;
        const int j = p - i * NN;
        // A tile: lanes 0-15 -> enc[i] in K0-7, dist in K16; lanes 16-31 -> enc[j] in K8-15.
        const float* src = enc + (hf ? j : i) * GAT_OUT;
        v16h a;
#pragma unroll
        for (int t = 0; t < 8; ++t) a[t] = (_Float16)src[t];
        a[8] = hf ? (_Float16)0.f : (_Float16)dist[p];
#pragma unroll
        for (int t = 9; t < 16; ++t) a[t] = (_Float16)0.f;

        float acc[8];
#pragma unroll
        for (int r = 0; r < 8; ++r) acc[r] = 0.f;

#pragma unroll
        for (int c = 0; c < 8; ++c) {
            v8f d = {};
            d = __builtin_amdgcn_wmma_f32_16x16x32_f16(
                    false, a, false, Breg[c], (short)0, d, false, false);
#pragma unroll
            for (int r = 0; r < 8; ++r) {
                float hv = d[r] + bias[c];
                hv = (hv > 0.f) ? hv : 0.f;     // relu
                acc[r] += hv * w2[c];           // fold second layer (rank-1)
            }
        }
        // Reduce over the 16 hidden columns held across lanes of each half.
#pragma unroll
        for (int r = 0; r < 8; ++r) {
            float v = acc[r];
            v += __shfl_xor(v, 1, 32);
            v += __shfl_xor(v, 2, 32);
            v += __shfl_xor(v, 4, 32);
            v += __shfl_xor(v, 8, 32);
            acc[r] = v;
        }
        if (l == 0) {
            const int rowBase = tile * 16 + hf * 8;  // D rows: VGPR r -> M=r (+8 for hi half)
#pragma unroll
            for (int r = 0; r < 8; ++r) out[rowBase + r] = acc[r] + b2;
        }
    }
}

// ---------------------------------------------------------------------------
extern "C" void kernel_launch(void* const* d_in, const int* in_sizes, int n_in,
                              void* d_out, int out_size, void* d_ws, size_t ws_size,
                              hipStream_t stream) {
    const float* geo      = (const float*)d_in[0];
    const float* sem      = (const float*)d_in[1];
    const float* features = (const float*)d_in[2];
    // d_in[3] region_pairs: identity meshgrid -> i = p/N, j = p%N, not needed
    const float* dist     = (const float*)d_in[4];
    const float* W0  = (const float*)d_in[5];
    const float* W1  = (const float*)d_in[6];
    const float* W2  = (const float*)d_in[7];
    const float* a0  = (const float*)d_in[8];
    const float* a1  = (const float*)d_in[9];
    const float* a2  = (const float*)d_in[10];
    const float* fc1_w = (const float*)d_in[11];
    const float* fc1_b = (const float*)d_in[12];
    const float* fc2_w = (const float*)d_in[13];
    const float* fc2_b = (const float*)d_in[14];
    float* out = (float*)d_out;

    float* ws   = (float*)d_ws;          // ~156 KB of f32 scratch
    float* encA = ws;                    // 1500*8
    float* encB = ws + 12000;            // 1500*8
    float* hbuf = ws + 24000;            // 1500*8
    float* fsrc = ws + 36000;            // 1500
    float* fdst = ws + 37500;            // 1500

    const int projBlocks = (NN + 255) / 256;

    // Layer 0 (K=32): features -> encA
    gat_proj<<<projBlocks, 256, 0, stream>>>(features, GAT_IN, W0, a0, hbuf, fsrc, fdst);
    gat_attn<<<NN, 256, 0, stream>>>(geo, sem, hbuf, fsrc, fdst, encA);
    // Layer 1 (K=8): encA -> encB
    gat_proj<<<projBlocks, 256, 0, stream>>>(encA, GAT_OUT, W1, a1, hbuf, fsrc, fdst);
    gat_attn<<<NN, 256, 0, stream>>>(geo, sem, hbuf, fsrc, fdst, encB);
    // Layer 2 (K=8): encB -> encA
    gat_proj<<<projBlocks, 256, 0, stream>>>(encB, GAT_OUT, W2, a2, hbuf, fsrc, fdst);
    gat_attn<<<NN, 256, 0, stream>>>(geo, sem, hbuf, fsrc, fdst, encA);

    // Fused edge MLP over all N^2 pairs (exactly 140625 tiles of 16, no tail)
    const int numTiles = (NN * NN) / 16;
    edge_mlp_wmma<<<2048, 256, 0, stream>>>(encA, dist, fc1_w, fc1_b, fc2_w, fc2_b,
                                            out, numTiles);
}
